// DualRNNDecoder_48842368090415
// MI455X (gfx1250) — matrix operand order repaired and checked by reading, not compiled, not run on hardware
//
#include <hip/hip_runtime.h>
#include <hip/hip_bf16.h>

typedef __attribute__((ext_vector_type(16))) _Float16     v16h;
typedef __attribute__((ext_vector_type(8)))  float        v8f;
typedef __attribute__((ext_vector_type(8)))  unsigned int v8u;

static __device__ __forceinline__ v8f wmma_f16f32(v16h a, v16h b, v8f c) {
  // D = A(16x32 f16) * B(32x16 f16) + C(16x16 f32)
  return __builtin_amdgcn_wmma_f32_16x16x32_f16(false, a, false, b, (short)0, c,
                                                false, false);
}

static __device__ __forceinline__ v8f zero8() {
  v8f z = {0.f, 0.f, 0.f, 0.f, 0.f, 0.f, 0.f, 0.f};
  return z;
}

static __device__ __forceinline__ float sigm(float x) {
  return 1.0f / (1.0f + __expf(-x));
}

// ---------------------------------------------------------------------------
// A fragment: 16x32 f16 tile, row-major source (ld halfs per row, even).
// lane m=l%16, half-group g=l/16;
//   VGPR v 0..7 : K = kbase + 16g + 2v, +1
// One contiguous 32B run -> 2x (ds_load / global_load) b128.
static __device__ __forceinline__ v16h load_a_frag(const _Float16* src, int ld,
                                                   int kbase, int lane) {
  int m = lane & 15, g = (lane >> 4) & 1;
  const unsigned int* p = (const unsigned int*)src;
  int base = (m * ld + kbase + 16 * g) >> 1;  // u32 units
  v8u t;
  t[0] = p[base + 0];
  t[1] = p[base + 1];
  t[2] = p[base + 2];
  t[3] = p[base + 3];
  t[4] = p[base + 4];
  t[5] = p[base + 5];
  t[6] = p[base + 6];
  t[7] = p[base + 7];
  return __builtin_bit_cast(v16h, t);
}

// B fragment: 32x16 f16 tile from a COLUMN-MAJOR strip: bt[n][k], n relative
// to the tile, K halfs per column. lane n=l%16, K-group g=l/16;
// VGPR v: K = kbase + 16g + 2v, +1. One contiguous 32B run -> 2x b128 loads.
// Works for both global (pre-transposed weights) and LDS staging.
static __device__ __forceinline__ v16h load_b_frag_ct(const _Float16* bt, int K,
                                                      int kbase, int lane) {
  int n = lane & 15, g = (lane >> 4) & 1;
  const unsigned int* p = (const unsigned int*)bt;
  int base = (n * K + kbase + 16 * g) >> 1;
  v8u t;
#pragma unroll
  for (int v = 0; v < 8; ++v) t[v] = p[base + v];
  return __builtin_bit_cast(v16h, t);
}

// ---------------------------------------------------------------------------
// One-time tiled transpose + f32->f16: src[R][C] -> dst[C][R].
__global__ void __launch_bounds__(256)
transpose_f32_to_f16_kernel(const float* __restrict__ src,
                            _Float16* __restrict__ dst, int R, int C) {
  __shared__ float tile[32][33];
  int c0 = blockIdx.x * 32;
  int r0 = blockIdx.y * 32;
  for (int dr = threadIdx.y; dr < 32; dr += blockDim.y) {
    int r = r0 + dr, c = c0 + threadIdx.x;
    if (r < R && c < C) tile[dr][threadIdx.x] = src[(long)r * C + c];
  }
  __syncthreads();
  for (int dc = threadIdx.y; dc < 32; dc += blockDim.y) {
    int c = c0 + dc, r = r0 + threadIdx.x;
    if (c < C && r < R) dst[(long)c * R + r] = (_Float16)tile[threadIdx.x][dc];
  }
}

__global__ void embed_kernel(const int* __restrict__ ids,
                             const float* __restrict__ emb,
                             _Float16* __restrict__ dst, long BT, long E) {
  long n = BT * E;
  long stride = (long)gridDim.x * blockDim.x;
  for (long i = (long)blockIdx.x * blockDim.x + threadIdx.x; i < n; i += stride) {
    long row = i / E, k = i - row * E;
    dst[i] = (_Float16)emb[(long)ids[row] * E + k];
  }
}

// ---------------------------------------------------------------------------
// Persistent stacked-LSTM: one workgroup (32 wave32), loops over T steps.
//   z = x @ W + h @ U + b ; gates i,f,g,o split along N (4*HID columns).
// Weights are PRE-TRANSPOSED f16: WT[4*HID][K] column strips, so each B
// fragment is 2x b128 global loads. All 4 gate B fragments are loaded before
// the 4 WMMAs so the loads clause together under one wait and the WMMAs run
// back-to-back. Wave w owns column blocks j = w, w+32, ...
// Dyn LDS: hbuf[2][16][HID] | xbuf[16][HID] | ybuf[16][HID], all f16.
__global__ void __launch_bounds__(1024)
lstm_kernel(const _Float16* __restrict__ x_seq,  // [(b*T+t)][IN0] f16
            const _Float16* __restrict__ W0t,    // [4H][IN0]
            const _Float16* __restrict__ U0t,    // [4H][HID]
            const float* __restrict__ b0,
            const _Float16* __restrict__ W1t,    // [4H][HID]
            const _Float16* __restrict__ U1t,    // [4H][HID]
            const float* __restrict__ b1,
            const float* __restrict__ initA,     // [2][16][512]
            const float* __restrict__ srcHB,     // nullable [2][16][512]
            const float* __restrict__ srcCB,     // nullable [2][16][512]
            float* __restrict__ cws,             // [2][16][HID] fp32 cell
            _Float16* __restrict__ out_f16,      // [(b*T+t)][HID]
            float* __restrict__ hfin,            // nullable [2][16][HID]
            float* __restrict__ cfin,            // nullable [2][16][HID]
            int IN0, int HID, int T) {
  extern __shared__ _Float16 smem[];
  _Float16* hbuf = smem;                   // 2*16*HID
  _Float16* xbuf = smem + 2 * 16 * HID;    // 16*HID
  _Float16* ybuf = xbuf + 16 * HID;        // 16*HID

  const int tid = threadIdx.x;
  const int lane = tid & 31;
  const int wave = tid >> 5;
  const int nwaves = blockDim.x >> 5;
  const int nblocks = HID >> 4;

  // ---- initial states ------------------------------------------------------
  for (int idx = tid; idx < 2 * 16 * HID; idx += blockDim.x) {
    int l = idx / (16 * HID);
    int r = idx - l * 16 * HID;
    int m = r / HID, n = r - m * HID;
    float hv, cv;
    if (srcHB == nullptr) {  // RNN1: h = c = init_state[l]
      hv = initA[(l * 16 + m) * 512 + n];
      cv = hv;
    } else {                 // RNN2: concat(init_state[l], rnn1 final)
      if (n < 512) {
        hv = initA[(l * 16 + m) * 512 + n];
        cv = hv;
      } else {
        hv = srcHB[(l * 16 + m) * 512 + (n - 512)];
        cv = srcCB[(l * 16 + m) * 512 + (n - 512)];
      }
    }
    hbuf[idx] = (_Float16)hv;
    cws[idx] = cv;
  }
  __syncthreads();

  for (int t = 0; t < T; ++t) {
    for (int l = 0; l < 2; ++l) {
      const int IN = (l == 0) ? IN0 : HID;
      const _Float16* Wt = l ? W1t : W0t;
      const _Float16* Ut = l ? U1t : U0t;
      const float* bias = l ? b1 : b0;

      if (l == 0) {  // stage embedding rows into LDS (u32-wide copies)
        const unsigned int* xs = (const unsigned int*)x_seq;
        unsigned int* xb = (unsigned int*)xbuf;
        int INw = IN0 >> 1;
        for (int idx = tid; idx < 16 * INw; idx += blockDim.x) {
          int m = idx / INw, k = idx - m * INw;
          xb[idx] = xs[((long)m * T + t) * INw + k];
        }
        __syncthreads();
      }
      const _Float16* hA = hbuf + l * 16 * HID;

      for (int j = wave; j < nblocks; j += nwaves) {
        v8f acc[4];
#pragma unroll
        for (int q = 0; q < 4; ++q) acc[q] = zero8();

        {  // x @ W
          const _Float16* bt0 = Wt + (long)(0 * HID + j * 16) * IN;
          const _Float16* bt1 = Wt + (long)(1 * HID + j * 16) * IN;
          const _Float16* bt2 = Wt + (long)(2 * HID + j * 16) * IN;
          const _Float16* bt3 = Wt + (long)(3 * HID + j * 16) * IN;
          for (int kb = 0; kb < IN; kb += 32) {
            v16h a  = load_a_frag(xbuf, IN, kb, lane);
            v16h f0 = load_b_frag_ct(bt0, IN, kb, lane);
            v16h f1 = load_b_frag_ct(bt1, IN, kb, lane);
            v16h f2 = load_b_frag_ct(bt2, IN, kb, lane);
            v16h f3 = load_b_frag_ct(bt3, IN, kb, lane);
            acc[0] = wmma_f16f32(a, f0, acc[0]);
            acc[1] = wmma_f16f32(a, f1, acc[1]);
            acc[2] = wmma_f16f32(a, f2, acc[2]);
            acc[3] = wmma_f16f32(a, f3, acc[3]);
          }
        }
        {  // h @ U
          const _Float16* bt0 = Ut + (long)(0 * HID + j * 16) * HID;
          const _Float16* bt1 = Ut + (long)(1 * HID + j * 16) * HID;
          const _Float16* bt2 = Ut + (long)(2 * HID + j * 16) * HID;
          const _Float16* bt3 = Ut + (long)(3 * HID + j * 16) * HID;
          for (int kb = 0; kb < HID; kb += 32) {
            v16h a  = load_a_frag(hA, HID, kb, lane);
            v16h f0 = load_b_frag_ct(bt0, HID, kb, lane);
            v16h f1 = load_b_frag_ct(bt1, HID, kb, lane);
            v16h f2 = load_b_frag_ct(bt2, HID, kb, lane);
            v16h f3 = load_b_frag_ct(bt3, HID, kb, lane);
            acc[0] = wmma_f16f32(a, f0, acc[0]);
            acc[1] = wmma_f16f32(a, f1, acc[1]);
            acc[2] = wmma_f16f32(a, f2, acc[2]);
            acc[3] = wmma_f16f32(a, f3, acc[3]);
          }
        }

        // epilogue: gates -> new c (fp32, ws) and new h (f16, staged in ybuf)
        int nn = lane & 15, g = (lane >> 4) & 1;
        int col = j * 16 + nn;
        float bi = bias[col];
        float bf = bias[HID + col];
        float bg = bias[2 * HID + col];
        float bo = bias[3 * HID + col];
        float* crow = cws + (long)l * 16 * HID;
#pragma unroll
        for (int e = 0; e < 8; ++e) {
          int m = e + 8 * g;
          float zi = acc[0][e] + bi;
          float zf = acc[1][e] + bf;
          float zg = acc[2][e] + bg;
          float zo = acc[3][e] + bo;
          long cidx = (long)m * HID + col;
          float c2 = sigm(zf) * crow[cidx] + sigm(zi) * tanhf(zg);
          float h2 = sigm(zo) * tanhf(c2);
          crow[cidx] = c2;
          ybuf[m * HID + col] = (_Float16)h2;
        }
      }
      __syncthreads();  // all GEMM reads of hbuf[l]/xbuf done before overwrite

      {  // publish new h: recurrent state + next-layer input (+ out for l==1)
        unsigned int* yb = (unsigned int*)ybuf;
        unsigned int* hb = (unsigned int*)(hbuf + l * 16 * HID);
        unsigned int* xb = (unsigned int*)xbuf;
        unsigned int* ob = (unsigned int*)out_f16;
        int HIDw = HID >> 1;
        for (int idx = tid; idx < 16 * HIDw; idx += blockDim.x) {
          unsigned int hv = yb[idx];
          hb[idx] = hv;
          xb[idx] = hv;
          if (l == 1) {
            int m = idx / HIDw, n2 = idx - m * HIDw;
            ob[((long)m * T + t) * HIDw + n2] = hv;
          }
        }
      }
      if (t == T - 1 && hfin != nullptr) {
        for (int idx = tid; idx < 16 * HID; idx += blockDim.x) {
          hfin[(long)l * 16 * HID + idx] = (float)ybuf[idx];
          cfin[(long)l * 16 * HID + idx] = cws[(long)l * 16 * HID + idx];
        }
      }
      __syncthreads();
    }
  }
}

// ---------------------------------------------------------------------------
// logits[r, n0:n0+16] = A[r,:] @ Wp[:, n0:n0+16] + bp.
// One block per 16-column strip: Wp strip converted f32->f16 into LDS once
// (Wp streamed from HBM exactly once), 8 waves sweep the 64 M-tiles.
__global__ void __launch_bounds__(256)
proj_kernel(const _Float16* __restrict__ A,  // [Mrows][K] f16
            const float* __restrict__ Wp,    // [K][NV] f32
            const float* __restrict__ bp,    // [NV]
            float* __restrict__ out,         // [Mrows][NV]
            int K, int NV, int Mrows) {
  extern __shared__ _Float16 sB[];  // transposed [16][K]
  const int n0 = blockIdx.x * 16;

  for (int idx = threadIdx.x; idx < 16 * K; idx += blockDim.x) {
    int k = idx >> 4, n = idx & 15;  // coalesced over n within a row of Wp
    sB[n * K + k] = (_Float16)Wp[(long)k * NV + n0 + n];
  }
  __syncthreads();

  const int lane = threadIdx.x & 31;
  const int wave = threadIdx.x >> 5;
  const int nw = blockDim.x >> 5;
  const int n = lane & 15;
  const int g = (lane >> 4) & 1;
  const float bias = bp[n0 + n];

  for (int mt = wave; mt < (Mrows >> 4); mt += nw) {
    v8f acc = zero8();
    const _Float16* Arow = A + (long)mt * 16 * K;
#pragma unroll 2
    for (int kb = 0; kb < K; kb += 32) {
      v16h a = load_a_frag(Arow, K, kb, lane);
      v16h b = load_b_frag_ct(sB, K, kb, lane);
      acc = wmma_f16f32(a, b, acc);
    }
#pragma unroll
    for (int e = 0; e < 8; ++e) {
      int m = mt * 16 + e + 8 * g;
      out[(long)m * NV + n0 + n] = acc[e] + bias;
    }
  }
}

// ---------------------------------------------------------------------------
extern "C" void kernel_launch(void* const* d_in, const int* in_sizes, int n_in,
                              void* d_out, int out_size, void* d_ws, size_t ws_size,
                              hipStream_t stream) {
  (void)in_sizes; (void)n_in; (void)out_size; (void)ws_size;
  const int B = 16, T = 64, V = 32000, E = 512, H = 512;
  const int BT = B * T;

  const int*   ids1 = (const int*)d_in[0];
  const int*   ids2 = (const int*)d_in[1];
  const float* init = (const float*)d_in[2];
  const float* emb  = (const float*)d_in[3];
  const float* W1_0 = (const float*)d_in[4];
  const float* U1_0 = (const float*)d_in[5];
  const float* b1_0 = (const float*)d_in[6];
  const float* W1_1 = (const float*)d_in[7];
  const float* U1_1 = (const float*)d_in[8];
  const float* b1_1 = (const float*)d_in[9];
  const float* W2_0 = (const float*)d_in[10];
  const float* U2_0 = (const float*)d_in[11];
  const float* b2_0 = (const float*)d_in[12];
  const float* W2_1 = (const float*)d_in[13];
  const float* U2_1 = (const float*)d_in[14];
  const float* b2_1 = (const float*)d_in[15];
  const float* Wp1  = (const float*)d_in[16];
  const float* bp1  = (const float*)d_in[17];
  const float* Wp2  = (const float*)d_in[18];
  const float* bp2  = (const float*)d_in[19];

  float* logits1 = (float*)d_out;
  float* logits2 = (float*)d_out + (long)BT * V;

  // ---- workspace layout (bytes, 256-aligned) ------------------------------
  char* ws = (char*)d_ws;
  size_t off = 0;
  auto alloc = [&](size_t bytes) -> char* {
    char* p = ws + off;
    off = (off + bytes + 255) & ~(size_t)255;
    return p;
  };
  _Float16* e1h   = (_Float16*)alloc((size_t)BT * E * 2);
  _Float16* e2h   = (_Float16*)alloc((size_t)BT * E * 2);
  _Float16* w10t  = (_Float16*)alloc((size_t)E * 4 * H * 2);      // [4H][E]
  _Float16* u10t  = (_Float16*)alloc((size_t)H * 4 * H * 2);      // [4H][H]
  _Float16* w11t  = (_Float16*)alloc((size_t)H * 4 * H * 2);
  _Float16* u11t  = (_Float16*)alloc((size_t)H * 4 * H * 2);
  _Float16* w20t  = (_Float16*)alloc((size_t)E * 8 * H * 2);      // [8H][E]
  _Float16* u20t  = (_Float16*)alloc((size_t)2 * H * 8 * H * 2);  // [8H][2H]
  _Float16* w21t  = (_Float16*)alloc((size_t)2 * H * 8 * H * 2);
  _Float16* u21t  = (_Float16*)alloc((size_t)2 * H * 8 * H * 2);
  _Float16* out1h = (_Float16*)alloc((size_t)BT * H * 2);
  _Float16* out2h = (_Float16*)alloc((size_t)BT * 2 * H * 2);
  float*    c1ws  = (float*)alloc((size_t)2 * 16 * H * 4);
  float*    c2ws  = (float*)alloc((size_t)2 * 16 * 2 * H * 4);
  float*    h1fin = (float*)alloc((size_t)2 * 16 * H * 4);
  float*    c1fin = (float*)alloc((size_t)2 * 16 * H * 4);

  // ---- 1) transpose-convert LSTM weights to f16 [N][K] --------------------
  dim3 tb(32, 8);
  auto tcvt = [&](const float* s, _Float16* d, int R, int C) {
    dim3 grid((C + 31) / 32, (R + 31) / 32);
    transpose_f32_to_f16_kernel<<<grid, tb, 0, stream>>>(s, d, R, C);
  };
  tcvt(W1_0, w10t, E, 4 * H);
  tcvt(U1_0, u10t, H, 4 * H);
  tcvt(W1_1, w11t, H, 4 * H);
  tcvt(U1_1, u11t, H, 4 * H);
  tcvt(W2_0, w20t, E, 8 * H);
  tcvt(U2_0, u20t, 2 * H, 8 * H);
  tcvt(W2_1, w21t, 2 * H, 8 * H);
  tcvt(U2_1, u21t, 2 * H, 8 * H);

  // ---- 2) embedding gathers ----------------------------------------------
  embed_kernel<<<2048, 256, 0, stream>>>(ids1, emb, e1h, BT, E);
  embed_kernel<<<2048, 256, 0, stream>>>(ids2, emb, e2h, BT, E);

  // ---- 3) RNN1 (HID=512): dyn LDS = 128*HID bytes -------------------------
  lstm_kernel<<<1, 1024, 128 * H, stream>>>(
      e1h, w10t, u10t, b1_0, w11t, u11t, b1_1,
      init, nullptr, nullptr, c1ws, out1h, h1fin, c1fin, E, H, T);

  // ---- 4) RNN2 (HID=1024), init = concat(init_state, rnn1 final) ----------
  lstm_kernel<<<1, 1024, 128 * 2 * H, stream>>>(
      e2h, w20t, u20t, b2_0, w21t, u21t, b2_1,
      init, h1fin, c1fin, c2ws, out2h, nullptr, nullptr, E, 2 * H, T);

  // ---- 5) projections: one block per 16-column strip of V -----------------
  proj_kernel<<<V / 16, 256, 32 * H, stream>>>(out1h, Wp1, bp1, logits1,
                                               H, V, BT);
  proj_kernel<<<V / 16, 256, 32 * 2 * H, stream>>>(out2h, Wp2, bp2, logits2,
                                                   2 * H, V, BT);
}
